// HybridContrastiveLoss_49684181680814
// MI455X (gfx1250) — matrix operand order repaired and checked by reading, not compiled, not run on hardware
//
#include <hip/hip_runtime.h>
#include <hip/hip_bf16.h>

typedef __attribute__((ext_vector_type(16))) _Float16 v16h;
typedef __attribute__((ext_vector_type(8)))  _Float16 v8h;
typedef __attribute__((ext_vector_type(8)))  float    v8f;

#define NROWS 8000
#define CDIM  256
#define HH 40
#define WW 40
#define SEL 4000
#define INV_TEMP 10.0f
#define REF_PT   10.0f   // fixed softmax reference: |dot/T| <= ~10.2 for unit vectors
#define NTILES   (NROWS / 16)      // 500 column tiles
#define LDSROW   264               // halves per LDS row: 256 + 8 pad (16B) -> bank-spread

// ---------------------------------------------------------------------------
// Kernel 1: convert anchors to f16, gather memory bank (NCHW -> [row][C] f16),
// memory labels, and pos[i] = dot(feat1_i, feat2_i) / TEMP in f32.
// grid: 8000 blocks x 256 threads (one block per row, one thread per channel)
// ---------------------------------------------------------------------------
__global__ void hcl_prep_kernel(const float* __restrict__ f1, const float* __restrict__ f2,
                                const float* __restrict__ ul1, const float* __restrict__ ul2,
                                const int* __restrict__ pl1, const int* __restrict__ pl2,
                                const int* __restrict__ sel1, const int* __restrict__ sel2,
                                _Float16* __restrict__ a1, _Float16* __restrict__ a2,
                                _Float16* __restrict__ mem,
                                int* __restrict__ memlab, float* __restrict__ pos)
{
    const int r = blockIdx.x;      // 0..7999
    const int c = threadIdx.x;     // 0..255
    const size_t idx = (size_t)r * CDIM + c;

    const float x1 = f1[idx];
    const float x2 = f2[idx];
    a1[idx] = (_Float16)x1;
    a2[idx] = (_Float16)x2;

    // block-wide dot product for pos[r]
    __shared__ float red[256];
    red[c] = x1 * x2;
    __syncthreads();
    #pragma unroll
    for (int s = 128; s > 0; s >>= 1) {
        if (c < s) red[c] += red[c + s];
        __syncthreads();
    }
    if (c == 0) pos[r] = red[0] * INV_TEMP;

    // memory bank row r: first SEL rows from ul1[sel1], rest from ul2[sel2]
    int src, which;
    if (r < SEL) { src = sel1[r];        which = 0; }
    else         { src = sel2[r - SEL];  which = 1; }
    const float* ul = which ? ul2 : ul1;
    const int b   = src / (HH * WW);
    const int rem = src % (HH * WW);
    // NCHW layout: ul[((b*C + c)*H*W) + rem]
    mem[idx] = (_Float16)ul[((size_t)b * CDIM + c) * (HH * WW) + rem];
    if (c == 0) memlab[r] = which ? pl2[src] : pl1[src];
}

// ---------------------------------------------------------------------------
// Kernel 2: fused GEMM (anchor @ memory.T) + masked exp-sum at fixed reference.
// grid: (125, 2) blocks x 128 threads (4 waves).  blockIdx.y selects anchor.
// Each wave owns a 16-row tile (block = 64 rows) and sweeps ALL 500 column
// tiles.  The 16x256 B panel is staged once per block into double-buffered
// LDS (coalesced global b128 -> ds_store), so all 4 waves feed their WMMAs
// from LDS instead of redundant L2 reads (4x L2 traffic reduction).
// Per row: m = max_j neg[i][j] (unmasked), s = sum_j mask * exp(neg[i][j]-10).
// Mask replicates the reference's transposed broadcast:
//   mask(i, j) = (anchor_labels[j] != memory_labels[i])
// ---------------------------------------------------------------------------
__global__ void __launch_bounds__(128)
hcl_gemm_softmax_kernel(const _Float16* __restrict__ A1, const _Float16* __restrict__ A2,
                        const _Float16* __restrict__ Mem,
                        const int* __restrict__ plab1, const int* __restrict__ plab2,
                        const int* __restrict__ memlab,
                        float* __restrict__ row_m, float* __restrict__ row_s)
{
    const int anchor = blockIdx.y;
    const _Float16* __restrict__ A    = anchor ? A2 : A1;
    const int*      __restrict__ alab = anchor ? plab2 : plab1;

    const int tid     = threadIdx.x;
    const int wave    = tid >> 5;        // 0..3 (wave32)
    const int lane    = tid & 31;
    const int rowBase = blockIdx.x * 64 + wave * 16;
    const int lmod    = lane & 15;
    const int lhi     = lane >> 4;       // 0 or 1

    // double-buffered B panel: 16 rows x 264 halves (8 pad halves -> bank spread)
    __shared__ _Float16 lds_b[2][16 * LDSROW];

    // ---- load the wave's full 16x256 A panel into registers (spec A layout) ----
    // lane<16 : halves[0..7]=K(kc*32+0..7),  halves[8..15]=K(kc*32+16..23)
    // lane>=16: halves[0..7]=K(kc*32+8..15), halves[8..15]=K(kc*32+24..31)
    v16h afrag[8];
    {
        const _Float16* ap = A + (size_t)(rowBase + lmod) * CDIM + lhi * 8;
        #pragma unroll
        for (int kc = 0; kc < 8; ++kc) {
            v8h lo = *(const v8h*)(ap + kc * 32);
            v8h hi = *(const v8h*)(ap + kc * 32 + 16);
            afrag[kc] = __builtin_shufflevector(lo, hi,
                0,1,2,3,4,5,6,7,8,9,10,11,12,13,14,15);
        }
    }

    // memory labels for the 8 rows this lane's accumulator slots cover
    // (C/D layout: slot r, lane group lhi -> row rowBase + r + 8*lhi)
    int mlab[8];
    #pragma unroll
    for (int r = 0; r < 8; ++r) mlab[r] = memlab[rowBase + r + lhi * 8];

    float m_run[8], s_run[8];
    #pragma unroll
    for (int r = 0; r < 8; ++r) { m_run[r] = -1e30f; s_run[r] = 0.0f; }

    // staging role: thread t loads 64B of the panel: row = t>>3, seg = t&7
    const int srow = tid >> 3;
    const int sseg = tid & 7;
    const _Float16* sgp0 = Mem + (size_t)srow * CDIM + sseg * 32;
    _Float16*       slp  = &lds_b[0][srow * LDSROW + sseg * 32];

    v8h stage0, stage1, stage2, stage3;

    // prologue: stage tile 0 into buffer 0
    {
        const _Float16* gp = sgp0;                     // tile 0
        stage0 = *(const v8h*)(gp +  0);
        stage1 = *(const v8h*)(gp +  8);
        stage2 = *(const v8h*)(gp + 16);
        stage3 = *(const v8h*)(gp + 24);
        *(v8h*)(slp +  0) = stage0;
        *(v8h*)(slp +  8) = stage1;
        *(v8h*)(slp + 16) = stage2;
        *(v8h*)(slp + 24) = stage3;
    }
    __syncthreads();

    for (int t = 0; t < NTILES; ++t) {
        const int buf = t & 1;

        // issue global loads for tile t+1 (in flight during this tile's WMMAs)
        const bool more = (t + 1) < NTILES;
        if (more) {
            const _Float16* gp = sgp0 + (size_t)(t + 1) * 16 * CDIM;
            __builtin_prefetch(gp + (size_t)16 * CDIM, 0, 1);   // tile t+2 -> L2
            stage0 = *(const v8h*)(gp +  0);
            stage1 = *(const v8h*)(gp +  8);
            stage2 = *(const v8h*)(gp + 16);
            stage3 = *(const v8h*)(gp + 24);
        }

        // ---- compute tile t from LDS ----
        const int aj = alab[t * 16 + lmod];   // anchor label at this COLUMN (reference quirk)

        const _Float16* lbp = &lds_b[buf][lmod * LDSROW + lhi * 8];
        v16h bfrag[8];
        #pragma unroll
        for (int kc = 0; kc < 8; ++kc) {
            // B layout (K x 16): lane n holds column n, K = lhi*8 + 0..7 and 16 + lhi*8 + 0..7
            v8h lo = *(const v8h*)(lbp + kc * 32);
            v8h hi = *(const v8h*)(lbp + kc * 32 + 16);
            bfrag[kc] = __builtin_shufflevector(lo, hi,
                0,1,2,3,4,5,6,7,8,9,10,11,12,13,14,15);
        }

        v8f acc = {};
        #pragma unroll
        for (int kc = 0; kc < 8; ++kc)
            acc = __builtin_amdgcn_wmma_f32_16x16x32_f16(
                false, afrag[kc], false, bfrag[kc], (short)0, acc, false, false);

        #pragma unroll
        for (int r = 0; r < 8; ++r) {
            const float v = acc[r] * INV_TEMP;
            m_run[r] = fmaxf(m_run[r], v);                       // unmasked row max
            const float e = __expf(fmaf(acc[r], INV_TEMP, -REF_PT));
            s_run[r] += (aj != mlab[r]) ? e : 0.0f;              // masked exp-sum
        }

        // commit tile t+1 into the other buffer
        if (more) {
            _Float16* lp = &lds_b[buf ^ 1][srow * LDSROW + sseg * 32];
            *(v8h*)(lp +  0) = stage0;
            *(v8h*)(lp +  8) = stage1;
            *(v8h*)(lp + 16) = stage2;
            *(v8h*)(lp + 24) = stage3;
        }
        __syncthreads();
    }

    // ---- reduce across the 16 column-lanes of each half-wave (max / add) ----
    #pragma unroll
    for (int r = 0; r < 8; ++r) {
        #pragma unroll
        for (int off = 1; off < 16; off <<= 1) {
            m_run[r] = fmaxf(m_run[r], __shfl_xor(m_run[r], off, 32));
            s_run[r] += __shfl_xor(s_run[r], off, 32);
        }
    }

    // each wave covers all columns for its 16 rows -> direct store
    if (lmod == 0) {
        #pragma unroll
        for (int r = 0; r < 8; ++r) {
            const int row = rowBase + lhi * 8 + r;
            row_m[anchor * NROWS + row] = m_run[r];
            row_s[anchor * NROWS + row] = s_run[r];   // exp-sum relative to REF_PT
        }
    }
}

// ---------------------------------------------------------------------------
// Kernel 3: fold in pos column, compute both masked mean NLL losses, sum.
// row_s is relative to REF_PT; convert exactly: sum_expM = row_s * exp(REF - M).
// ---------------------------------------------------------------------------
__global__ void hcl_finalize_kernel(const float* __restrict__ row_m, const float* __restrict__ row_s,
                                    const float* __restrict__ pos,
                                    const float* __restrict__ plog1, const float* __restrict__ plog2,
                                    float* __restrict__ out)
{
    const int tid = threadIdx.x;
    float c1 = 0.f, n1 = 0.f, c2 = 0.f, n2 = 0.f;
    for (int i = tid; i < NROWS; i += 256) {
        const float p = pos[i];
        {   // loss1: anchor = feat1
            const float m = row_m[i], s = row_s[i];
            const float M = fmaxf(m, p);
            const float st = s * __expf(REF_PT - M) + __expf(p - M);  // + pos col (mask=1)
            const float lg = __expf(p - M) / (st + 1e-8f);
            const float mk = (plog2[i] > 0.7f && plog1[i] < plog2[i]) ? 1.f : 0.f;
            c1 += -__logf(lg + 1e-8f) * mk;  n1 += mk;
        }
        {   // loss2: anchor = feat2
            const float m = row_m[NROWS + i], s = row_s[NROWS + i];
            const float M = fmaxf(m, p);
            const float st = s * __expf(REF_PT - M) + __expf(p - M);
            const float lg = __expf(p - M) / (st + 1e-8f);
            const float mk = (plog1[i] > 0.7f && plog2[i] < plog1[i]) ? 1.f : 0.f;
            c2 += -__logf(lg + 1e-8f) * mk;  n2 += mk;
        }
    }
    __shared__ float r1[256], r2[256], r3[256], r4[256];
    r1[tid] = c1; r2[tid] = n1; r3[tid] = c2; r4[tid] = n2;
    __syncthreads();
    #pragma unroll
    for (int s = 128; s > 0; s >>= 1) {
        if (tid < s) { r1[tid]+=r1[tid+s]; r2[tid]+=r2[tid+s]; r3[tid]+=r3[tid+s]; r4[tid]+=r4[tid+s]; }
        __syncthreads();
    }
    if (tid == 0)
        out[0] = r1[0] / (r2[0] + 1e-12f) + r3[0] / (r4[0] + 1e-12f);
}

// ---------------------------------------------------------------------------
extern "C" void kernel_launch(void* const* d_in, const int* in_sizes, int n_in,
                              void* d_out, int out_size, void* d_ws, size_t ws_size,
                              hipStream_t stream) {
    const float* f1    = (const float*)d_in[0];
    const float* f2    = (const float*)d_in[1];
    const float* ul1   = (const float*)d_in[2];
    const float* ul2   = (const float*)d_in[3];
    const int*   pl1   = (const int*)d_in[4];
    const int*   pl2   = (const int*)d_in[5];
    const float* plog1 = (const float*)d_in[6];
    const float* plog2 = (const float*)d_in[7];
    const int*   sel1  = (const int*)d_in[8];
    const int*   sel2  = (const int*)d_in[9];
    float* out = (float*)d_out;

    char* ws = (char*)d_ws;
    const size_t SZ_F16 = (size_t)NROWS * CDIM * sizeof(_Float16); // 4,096,000 B
    _Float16* a1     = (_Float16*)(ws);
    _Float16* a2     = (_Float16*)(ws + SZ_F16);
    _Float16* mem    = (_Float16*)(ws + 2 * SZ_F16);
    float*    pos    = (float*)   (ws + 3 * SZ_F16);
    int*      memlab = (int*)     (ws + 3 * SZ_F16 + NROWS * 4);
    float*    row_m  = (float*)   (ws + 3 * SZ_F16 + 2 * (size_t)NROWS * 4);
    float*    row_s  = (float*)   (ws + 3 * SZ_F16 + 4 * (size_t)NROWS * 4);
    // total workspace use: 3*4,096,000 + 6*32,000 = 12,480,000 bytes

    hcl_prep_kernel<<<NROWS, 256, 0, stream>>>(f1, f2, ul1, ul2, pl1, pl2, sel1, sel2,
                                               a1, a2, mem, memlab, pos);
    hcl_gemm_softmax_kernel<<<dim3(NROWS / 64, 2), 128, 0, stream>>>(
        a1, a2, mem, pl1, pl2, memlab, row_m, row_s);
    hcl_finalize_kernel<<<1, 256, 0, stream>>>(row_m, row_s, pos, plog1, plog2, out);
}
